// TransformerTransferGSCPredictor_36764920054333
// MI455X (gfx1250) — compile-verified, matrix operand order checked by reading, not run on hardware
//
#include <hip/hip_runtime.h>
#include <hip/hip_bf16.h>

// ---------------------------------------------------------------------------
// MI455X (gfx1250): bf16 WMMA GEMM pipeline, f32 accumulate, TDM tile feeds.
// ---------------------------------------------------------------------------

typedef __attribute__((ext_vector_type(16))) __bf16 v16bf;
typedef __attribute__((ext_vector_type(8)))  float  v8f;

#define GB 16
#define GO 80
#define GR 512
#define GPOOL 4096
#define GNH 8
#define GDK 64
#define GNOBJ 151
#define GNREL 51

#define BM 128
#define BN 128
#define BK 32
#define LDSS 40   // padded LDS row stride (elements): banks 20*m mod 64 all-distinct

#if defined(__gfx1250__) && __has_builtin(__builtin_amdgcn_tensor_load_to_lds)
#define USE_TDM 1
#else
#define USE_TDM 0
#endif

#if __has_include(<hip/amd_detail/amd_gfx1250_TDM.h>)
#define TDM_6ARG 1
#else
#define TDM_6ARG 0
#endif

#if USE_TDM
typedef unsigned int u32x4 __attribute__((ext_vector_type(4)));
typedef int          i32x4 __attribute__((ext_vector_type(4)));
typedef int          i32x8 __attribute__((ext_vector_type(8)));

// Issue one TDM 2D tile load: `rows` rows of 32 bf16 elements, global row
// stride = row_elems*2 bytes, into LDS at lds_off with 16B pad per 64B row
// (matches the LDSS=40-element padded layout).
__device__ __forceinline__ void tdm_load_tile(const __bf16* gptr, unsigned lds_off,
                                              int rows, int row_elems) {
  unsigned long long ga = (unsigned long long)(const void*)gptr;
  u32x4 g0;
  g0[0] = 1u;                                        // count=1 (valid), user mode
  g0[1] = lds_off;                                   // LDS byte address
  g0[2] = (unsigned)(ga & 0xFFFFFFFFu);              // global addr lo
  g0[3] = (unsigned)((ga >> 32) & 0x01FFFFFFu) | (2u << 30);  // addr hi | type=2
  i32x8 g1;
  // wg_mask=0 | data_size=1(2B)<<16 | pad_en<<20 | pad_interval=3(16DW)<<22
  // | pad_amount=3(4DW)<<25
  g1[0] = (1 << 16) | (1 << 20) | (3 << 22) | (3 << 25);
  const unsigned td0 = 0x40000000u;                  // tensor_dim0 (large, no clip)
  const unsigned td1 = 0x40000000u;                  // tensor_dim1
  g1[1] = (int)((td0 & 0xFFFFu) << 16);              // [31:16]=tensor_dim0 lo16
  g1[2] = (int)(((td0 >> 16) & 0xFFFFu) | ((td1 & 0xFFFFu) << 16));
  g1[3] = (int)(((td1 >> 16) & 0xFFFFu) | (32u << 16));      // tile_dim0=32
  g1[4] = (int)((unsigned)rows & 0xFFFFu);           // tile_dim1=rows, tile_dim2=0
  unsigned long long st0 = (unsigned long long)(unsigned)row_elems;  // dim0 stride
  g1[5] = (int)(st0 & 0xFFFFFFFFu);
  g1[6] = (int)((st0 >> 32) & 0xFFFFu);              // stride hi16 | dim1_stride lo16=0
  g1[7] = 0;
  i32x4 z4 = {0, 0, 0, 0};
#if TDM_6ARG
  i32x8 z8 = {0, 0, 0, 0, 0, 0, 0, 0};
  __builtin_amdgcn_tensor_load_to_lds(g0, g1, z4, z4, z8, 0);
#else
  __builtin_amdgcn_tensor_load_to_lds(g0, g1, z4, z4, 0);
#endif
}
#endif  // USE_TDM

// ---------------------------------------------------------------------------
// f32 -> bf16 conversion (grid-stride)
// ---------------------------------------------------------------------------
__global__ void cvt_f32_bf16(const float* __restrict__ in, __bf16* __restrict__ out,
                             long long n) {
  long long i = (long long)blockIdx.x * blockDim.x + threadIdx.x;
  long long stride = (long long)gridDim.x * blockDim.x;
  for (; i < n; i += stride) out[i] = (__bf16)in[i];
}

// ---------------------------------------------------------------------------
// Transposing f32 -> bf16 weight conversion: out[n*K+k] = (bf16) in[k*N+n]
// 32x32 LDS tile; grids (N/32, K/32). One-shot per launch (weights only).
// ---------------------------------------------------------------------------
__global__ __launch_bounds__(256)
void transpose_cvt(const float* __restrict__ in, __bf16* __restrict__ out,
                   int K, int N) {
  __shared__ float t[32][33];
  const int k0 = blockIdx.y * 32, n0 = blockIdx.x * 32;
  const int tx = threadIdx.x & 31, ty = threadIdx.x >> 5;  // 8 row-groups
  for (int r = ty; r < 32; r += 8)
    t[r][tx] = in[(size_t)(k0 + r) * N + n0 + tx];
  __syncthreads();
  for (int r = ty; r < 32; r += 8)
    out[(size_t)(n0 + r) * K + k0 + tx] = (__bf16)t[tx][r];
}

// ---------------------------------------------------------------------------
// Tiled bf16 WMMA GEMM:  C[M,N] = A[M,K] @ B[K,N]  (+bias +residual, relu, *gate)
// A row-major [M,K] bf16, Bt = B TRANSPOSED row-major [N,K] bf16.
// 256 threads = 8 wave32 (4x2 wave grid), 128x128x32 block tile,
// double-buffered LDS fed by the Tensor Data Mover (fallback: manual loads).
// ---------------------------------------------------------------------------
__global__ __launch_bounds__(256)
void gemm_bf16_wmma(const __bf16* __restrict__ A, const __bf16* __restrict__ Bt,
                    const float* __restrict__ bias,
                    const float* __restrict__ residual,
                    const float* __restrict__ gate,
                    float* __restrict__ Cf, __bf16* __restrict__ Cb,
                    int M, int N, int K, int relu_flag) {
  __shared__ __bf16 sA[2][BM * LDSS];
  __shared__ __bf16 sB[2][BN * LDSS];

  const int tid  = threadIdx.x;
  const int bm   = blockIdx.y * BM;
  const int bn   = blockIdx.x * BN;
  const int wave = tid >> 5;
  const int lane = tid & 31;
  const int wm   = (wave & 3) * 32;
  const int wn   = (wave >> 2) * 64;

  v8f acc[2][4];
#pragma unroll
  for (int mt = 0; mt < 2; ++mt)
#pragma unroll
    for (int nt = 0; nt < 4; ++nt)
      acc[mt][nt] = (v8f){0.f, 0.f, 0.f, 0.f, 0.f, 0.f, 0.f, 0.f};

#if USE_TDM
  auto issue_tile = [&](int kt, int buf) {
    // A tile: BM rows x BK cols @ row stride K; Bt tile: BN rows x BK cols.
    tdm_load_tile(A + (size_t)bm * K + (size_t)kt * BK,
                  (unsigned)(unsigned long long)(const void*)&sA[buf][0], BM, K);
    tdm_load_tile(Bt + (size_t)bn * K + (size_t)kt * BK,
                  (unsigned)(unsigned long long)(const void*)&sB[buf][0], BN, K);
  };
#else
  const int l_row = tid >> 1;          // 0..127
  const int l_col = (tid & 1) * 16;    // 0 or 16
  auto load_tile = [&](int kt, int buf) {
    const __bf16* ap = A + (size_t)(bm + l_row) * K + (size_t)kt * BK + l_col;
    uint4 av0 = ((const uint4*)ap)[0];
    uint4 av1 = ((const uint4*)ap)[1];
    __bf16* sa = &sA[buf][l_row * LDSS + l_col];
    ((uint4*)sa)[0] = av0;
    ((uint4*)sa)[1] = av1;
    const __bf16* bp = Bt + (size_t)(bn + l_row) * K + (size_t)kt * BK + l_col;
    uint4 bv0 = ((const uint4*)bp)[0];
    uint4 bv1 = ((const uint4*)bp)[1];
    __bf16* sb = &sB[buf][l_row * LDSS + l_col];
    ((uint4*)sb)[0] = bv0;
    ((uint4*)sb)[1] = bv1;
  };
#endif

  auto compute = [&](int buf) {
    const int fm = lane & 15;
    const int kb = (lane >> 4) << 3;  // 0 or 8 (16-bit A/B WMMA VGPR layout)
    v16bf afrag[2], bfrag[4];
#pragma unroll
    for (int mt = 0; mt < 2; ++mt) {
      const __bf16* p = &sA[buf][(wm + mt * 16 + fm) * LDSS + kb];
      ((uint4*)&afrag[mt])[0] = *(const uint4*)p;
      ((uint4*)&afrag[mt])[1] = *(const uint4*)(p + 16);
    }
#pragma unroll
    for (int nt = 0; nt < 4; ++nt) {
      const __bf16* p = &sB[buf][(wn + nt * 16 + fm) * LDSS + kb];
      ((uint4*)&bfrag[nt])[0] = *(const uint4*)p;
      ((uint4*)&bfrag[nt])[1] = *(const uint4*)(p + 16);
    }
#pragma unroll
    for (int mt = 0; mt < 2; ++mt)
#pragma unroll
      for (int nt = 0; nt < 4; ++nt)
        acc[mt][nt] = __builtin_amdgcn_wmma_f32_16x16x32_bf16(
            false, afrag[mt], false, bfrag[nt], (short)0, acc[mt][nt], false, false);
  };

  const int KT = K / BK;
#if USE_TDM
  if (wave == 0) {
    issue_tile(0, 0);
    __builtin_amdgcn_s_wait_tensorcnt(0);
  }
  __syncthreads();
  for (int kt = 0; kt < KT; ++kt) {
    const int buf = kt & 1;
    if (wave == 0 && kt + 1 < KT) issue_tile(kt + 1, buf ^ 1);  // DMA next tile
    compute(buf);
    if (wave == 0) __builtin_amdgcn_s_wait_tensorcnt(0);
    __syncthreads();
  }
#else
  load_tile(0, 0);
  __syncthreads();
  for (int kt = 0; kt < KT; ++kt) {
    const int buf = kt & 1;
    if (kt + 1 < KT) load_tile(kt + 1, buf ^ 1);
    if (kt + 2 < KT) {
      __builtin_prefetch(A + (size_t)(bm + l_row) * K + (size_t)(kt + 2) * BK + l_col, 0, 1);
      __builtin_prefetch(Bt + (size_t)(bn + l_row) * K + (size_t)(kt + 2) * BK + l_col, 0, 1);
    }
    compute(buf);
    __syncthreads();
  }
#endif

  // Epilogue. C/D layout: VGPR j -> M = j (lanes 0-15) or j+8 (lanes 16-31); N = lane%16.
  const int cm = (lane >> 4) * 8;
#pragma unroll
  for (int mt = 0; mt < 2; ++mt) {
#pragma unroll
    for (int nt = 0; nt < 4; ++nt) {
      const int col = bn + wn + nt * 16 + (lane & 15);
      const float bv = bias ? bias[col] : 0.f;
#pragma unroll
      for (int j = 0; j < 8; ++j) {
        const int row = bm + wm + mt * 16 + cm + j;
        size_t idx = (size_t)row * N + col;
        float v = acc[mt][nt][j] + bv;
        if (residual) v += residual[idx];
        if (relu_flag) v = fmaxf(v, 0.f);
        if (gate) v *= gate[idx];
        if (Cf) Cf[idx] = v;
        if (Cb) Cb[idx] = (__bf16)v;
      }
    }
  }
}

// ---------------------------------------------------------------------------
// Gather head/tail object reps into prod_rep (f32 + bf16).
// ---------------------------------------------------------------------------
__global__ __launch_bounds__(256)
void build_prod(const float* __restrict__ edge_rep, const int* __restrict__ pairs,
                float* __restrict__ pf, __bf16* __restrict__ pb) {
  const int row = blockIdx.x;  // 0..8191
  const int b = row >> 9;
  const int is = pairs[row * 2 + 0];
  const int io = pairs[row * 2 + 1];
  const float* head = edge_rep + (size_t)(b * GO + is) * 1024;
  const float* tail = edge_rep + (size_t)(b * GO + io) * 1024;
  for (int c = threadIdx.x; c < 1024; c += 256) {
    float v = (c < 512) ? head[c] : tail[c];
    pf[(size_t)row * 1024 + c] = v;
    pb[(size_t)row * 1024 + c] = (__bf16)v;
  }
}

// ---------------------------------------------------------------------------
// Fused attention for one (batch, head): scores -> softmax -> attn@V.
// K,V per head = 64KB bf16 each -> L2-resident and reused 512x.
// ---------------------------------------------------------------------------
__global__ __launch_bounds__(256)
void attention(const __bf16* __restrict__ Q, const __bf16* __restrict__ Kb,
               const __bf16* __restrict__ V, __bf16* __restrict__ AO) {
  __shared__ float qs[64];
  __shared__ float sc[512];
  __shared__ float red[8];
  __shared__ float aop[4][64];
  const int bh = blockIdx.x;  // 0..127
  const int b = bh >> 3, h = bh & 7;
  const size_t base = (size_t)b * GR * (GNH * GDK) + (size_t)h * GDK;
  const int tid = threadIdx.x;
  const float scale = 0.125f;

  for (int q = 0; q < GR; ++q) {
    if (tid < 64) qs[tid] = (float)Q[base + (size_t)q * (GNH * GDK) + tid];
    __syncthreads();
    float lmax = -1e30f;
    for (int kk = tid; kk < GR; kk += 256) {
      const __bf16* kp = Kb + base + (size_t)kk * (GNH * GDK);
      float s = 0.f;
#pragma unroll 8
      for (int d = 0; d < GDK; ++d) s += qs[d] * (float)kp[d];
      s *= scale;
      sc[kk] = s;
      lmax = fmaxf(lmax, s);
    }
    for (int off = 16; off; off >>= 1) lmax = fmaxf(lmax, __shfl_down(lmax, off));
    if ((tid & 31) == 0) red[tid >> 5] = lmax;
    __syncthreads();
    if (tid == 0) {
      float m = red[0];
      for (int i = 1; i < 8; ++i) m = fmaxf(m, red[i]);
      red[0] = m;
    }
    __syncthreads();
    const float m = red[0];
    __syncthreads();
    float lsum = 0.f;
    for (int kk = tid; kk < GR; kk += 256) {
      float e = __expf(sc[kk] - m);
      sc[kk] = e;
      lsum += e;
    }
    for (int off = 16; off; off >>= 1) lsum += __shfl_down(lsum, off);
    if ((tid & 31) == 0) red[tid >> 5] = lsum;
    __syncthreads();
    if (tid == 0) {
      float s = 0.f;
      for (int i = 0; i < 8; ++i) s += red[i];
      red[0] = s;
    }
    __syncthreads();
    const float inv = 1.f / red[0];
    const int d = tid & 63, part = tid >> 6;
    float a = 0.f;
    const int k0 = part * 128;
    for (int kk = k0; kk < k0 + 128; ++kk)
      a += sc[kk] * (float)V[base + (size_t)kk * (GNH * GDK) + d];
    aop[part][d] = a;
    __syncthreads();
    if (tid < 64) {
      float r = (aop[0][tid] + aop[1][tid] + aop[2][tid] + aop[3][tid]) * inv;
      AO[base + (size_t)q * (GNH * GDK) + tid] = (__bf16)r;
    }
    __syncthreads();
  }
}

// ---------------------------------------------------------------------------
// Row layernorm: out = (x-mean)*rsqrt(var+eps)*g + b; f32 and/or bf16 out.
// ---------------------------------------------------------------------------
__global__ __launch_bounds__(256)
void layer_norm(const float* __restrict__ in, const float* __restrict__ g,
                const float* __restrict__ be, float* __restrict__ out_f,
                __bf16* __restrict__ out_b, int cols) {
  __shared__ float r1[8], r2[8];
  const int tid = threadIdx.x;
  const float* x = in + (size_t)blockIdx.x * cols;
  float s = 0.f, s2 = 0.f;
  for (int c = tid; c < cols; c += 256) {
    float v = x[c];
    s += v;
    s2 += v * v;
  }
  for (int off = 16; off; off >>= 1) {
    s += __shfl_down(s, off);
    s2 += __shfl_down(s2, off);
  }
  if ((tid & 31) == 0) { r1[tid >> 5] = s; r2[tid >> 5] = s2; }
  __syncthreads();
  if (tid == 0) {
    float a = 0.f, b2 = 0.f;
    for (int i = 0; i < 8; ++i) { a += r1[i]; b2 += r2[i]; }
    r1[0] = a; r2[0] = b2;
  }
  __syncthreads();
  const float mean = r1[0] / cols;
  const float var  = r2[0] / cols - mean * mean;
  const float inv  = rsqrtf(var + 1e-5f);
  for (int c = tid; c < cols; c += 256) {
    float v = (x[c] - mean) * inv * g[c] + be[c];
    size_t idx = (size_t)blockIdx.x * cols + c;
    if (out_f) out_f[idx] = v;
    if (out_b) out_b[idx] = (__bf16)v;
  }
}

// ---------------------------------------------------------------------------
// Output head: rel_dists = visual@W_rel + b_rel + prod@W_ctx + b_ctx + freq
// ---------------------------------------------------------------------------
__global__ __launch_bounds__(256)
void final_proj(const float* __restrict__ visual, const float* __restrict__ prod,
                const float* __restrict__ Wrel, const float* __restrict__ brel,
                const float* __restrict__ Wctx, const float* __restrict__ bctx,
                const float* __restrict__ freq, const int* __restrict__ pairs,
                const int* __restrict__ objp, float* __restrict__ out) {
  __shared__ float vrow[GPOOL];
  __shared__ float prow[1024];
  __shared__ float part[GNREL][4];
  const int row = blockIdx.x;  // 0..8191
  const int b = row >> 9;
  const int tid = threadIdx.x;
  for (int c = tid; c < GPOOL; c += 256) vrow[c] = visual[(size_t)row * GPOOL + c];
  for (int c = tid; c < 1024; c += 256) prow[c] = prod[(size_t)row * 1024 + c];
  __syncthreads();
  if (tid < GNREL * 4) {
    const int j = tid >> 2, p = tid & 3;
    float s = 0.f;
    for (int c = p; c < GPOOL; c += 4) s += vrow[c] * Wrel[(size_t)c * GNREL + j];
    for (int c = p; c < 1024; c += 4) s += prow[c] * Wctx[(size_t)c * GNREL + j];
    part[j][p] = s;
  }
  __syncthreads();
  if (tid < GNREL) {
    const int is = pairs[row * 2 + 0], io = pairs[row * 2 + 1];
    const int sub = objp[b * GO + is], ob = objp[b * GO + io];
    const float f = freq[((size_t)sub * GNOBJ + ob) * GNREL + tid];
    out[(size_t)row * GNREL + tid] =
        part[tid][0] + part[tid][1] + part[tid][2] + part[tid][3] +
        brel[tid] + bctx[tid] + f;
  }
}

// ---------------------------------------------------------------------------
// Host launcher
// ---------------------------------------------------------------------------
extern "C" void kernel_launch(void* const* d_in, const int* in_sizes, int n_in,
                              void* d_out, int out_size, void* d_ws, size_t ws_size,
                              hipStream_t stream) {
  (void)in_sizes; (void)n_in; (void)out_size; (void)ws_size;

  const float* edge_ctx   = (const float*)d_in[0];
  const float* union_feat = (const float*)d_in[1];
  const int*   pairs      = (const int*)d_in[2];
  const int*   obj_preds  = (const int*)d_in[3];
  const float* W_post_emb = (const float*)d_in[4];
  const float* b_post_emb = (const float*)d_in[5];
  const float* W_post_cat = (const float*)d_in[6];
  const float* b_post_cat = (const float*)d_in[7];
  const float* Wq  = (const float*)d_in[8];
  const float* Wk  = (const float*)d_in[9];
  const float* Wv  = (const float*)d_in[10];
  const float* Wo  = (const float*)d_in[11];
  const float* b_o = (const float*)d_in[12];
  const float* ln1_g = (const float*)d_in[13];
  const float* ln1_b = (const float*)d_in[14];
  const float* W_ff1 = (const float*)d_in[15];
  const float* b_ff1 = (const float*)d_in[16];
  const float* W_ff2 = (const float*)d_in[17];
  const float* b_ff2 = (const float*)d_in[18];
  const float* ln2_g = (const float*)d_in[19];
  const float* ln2_b = (const float*)d_in[20];
  const float* W_peu = (const float*)d_in[21];
  const float* b_peu = (const float*)d_in[22];
  const float* W_rel = (const float*)d_in[23];
  const float* b_rel = (const float*)d_in[24];
  const float* W_ctx = (const float*)d_in[25];
  const float* b_ctx = (const float*)d_in[26];
  const float* freq  = (const float*)d_in[27];
  float* out = (float*)d_out;

  const int MROWS = GB * GR;  // 8192
  const int EROWS = GB * GO;  // 1280

  char* w = (char*)d_ws;
  size_t off = 0;
  auto take = [&](size_t bytes) -> void* {
    void* p = w + off;
    off += (bytes + 255) & ~(size_t)255;
    return p;
  };
  // transposed bf16 weights [N,K]
  __bf16* Wpe_t  = (__bf16*)take((size_t)512 * 1024 * 2);
  __bf16* Wpc_t  = (__bf16*)take((size_t)1024 * 4096 * 2);
  __bf16* Wq_t   = (__bf16*)take((size_t)4096 * 512 * 2);
  __bf16* Wk_t   = (__bf16*)take((size_t)4096 * 512 * 2);
  __bf16* Wv_t   = (__bf16*)take((size_t)4096 * 512 * 2);
  __bf16* Wo_t   = (__bf16*)take((size_t)512 * 4096 * 2);
  __bf16* Wff1_t = (__bf16*)take((size_t)4096 * 2048 * 2);
  __bf16* Wff2_t = (__bf16*)take((size_t)2048 * 4096 * 2);
  __bf16* Wpeu_t = (__bf16*)take((size_t)4096 * 4096 * 2);
  // activations
  __bf16* ec_bf   = (__bf16*)take((size_t)EROWS * 512 * 2);
  float*  edge_rep= (float*) take((size_t)EROWS * 1024 * 4);
  float*  prod_f  = (float*) take((size_t)MROWS * 1024 * 4);
  __bf16* prod_bf = (__bf16*)take((size_t)MROWS * 1024 * 2);
  __bf16* x_bf    = (__bf16*)take((size_t)MROWS * 4096 * 2);
  float*  ctx_gate= (float*) take((size_t)MROWS * 4096 * 4);
  __bf16* q_bf    = (__bf16*)take((size_t)MROWS * 512 * 2);
  __bf16* k_bf    = (__bf16*)take((size_t)MROWS * 512 * 2);
  __bf16* v_bf    = (__bf16*)take((size_t)MROWS * 512 * 2);
  __bf16* ao_bf   = (__bf16*)take((size_t)MROWS * 512 * 2);
  float*  ybuf    = (float*) take((size_t)MROWS * 4096 * 4);
  float*  x1f     = (float*) take((size_t)MROWS * 4096 * 4);
  __bf16* x1_bf   = (__bf16*)take((size_t)MROWS * 4096 * 2);
  __bf16* ff_bf   = (__bf16*)take((size_t)MROWS * 2048 * 2);
  __bf16* uc_bf   = (__bf16*)take((size_t)MROWS * 4096 * 2);

  auto cvt = [&](const float* s, __bf16* d, long long n) {
    long long g = (n + 255) / 256;
    if (g > 8192) g = 8192;
    cvt_f32_bf16<<<dim3((unsigned)g), dim3(256), 0, stream>>>(s, d, n);
  };
  auto tcvt = [&](const float* s, __bf16* d, int K, int N) {
    transpose_cvt<<<dim3(N / 32, K / 32), dim3(256), 0, stream>>>(s, d, K, N);
  };
  auto gemm = [&](const __bf16* Ax, const __bf16* Bx, const float* bias,
                  const float* res, const float* gate, float* Cf, __bf16* Cb,
                  int M, int N, int K, int relu) {
    dim3 grid(N / BN, M / BM);
    gemm_bf16_wmma<<<grid, dim3(256), 0, stream>>>(Ax, Bx, bias, res, gate,
                                                   Cf, Cb, M, N, K, relu);
  };

  // 0) weight transpose+convert, activation convert
  tcvt(W_post_emb, Wpe_t, 512, 1024);
  tcvt(W_post_cat, Wpc_t, 1024, 4096);
  tcvt(Wq, Wq_t, 4096, 512);
  tcvt(Wk, Wk_t, 4096, 512);
  tcvt(Wv, Wv_t, 4096, 512);
  tcvt(Wo, Wo_t, 512, 4096);
  tcvt(W_ff1, Wff1_t, 4096, 2048);
  tcvt(W_ff2, Wff2_t, 2048, 4096);
  tcvt(W_peu, Wpeu_t, 4096, 4096);
  cvt(edge_ctx,   ec_bf, (long long)EROWS * 512);
  cvt(union_feat, x_bf,  (long long)MROWS * 4096);

  // 1) edge_rep = edge_ctx @ W_post_emb + b_post_emb
  gemm(ec_bf, Wpe_t, b_post_emb, nullptr, nullptr, edge_rep, nullptr,
       EROWS, 1024, 512, 0);

  // 2) gather -> prod_rep
  build_prod<<<dim3(MROWS), dim3(256), 0, stream>>>(edge_rep, pairs, prod_f, prod_bf);

  // 3) ctx_gate = prod_rep @ W_post_cat + b_post_cat
  gemm(prod_bf, Wpc_t, b_post_cat, nullptr, nullptr, ctx_gate, nullptr,
       MROWS, 4096, 1024, 0);

  // 4) Q,K,V projections (bf16 out)
  gemm(x_bf, Wq_t, nullptr, nullptr, nullptr, nullptr, q_bf, MROWS, 512, 4096, 0);
  gemm(x_bf, Wk_t, nullptr, nullptr, nullptr, nullptr, k_bf, MROWS, 512, 4096, 0);
  gemm(x_bf, Wv_t, nullptr, nullptr, nullptr, nullptr, v_bf, MROWS, 512, 4096, 0);

  // 5) fused attention
  attention<<<dim3(GB * GNH), dim3(256), 0, stream>>>(q_bf, k_bf, v_bf, ao_bf);

  // 6) y = union_feat + ao @ Wo + b_o
  gemm(ao_bf, Wo_t, b_o, union_feat, nullptr, ybuf, nullptr, MROWS, 4096, 512, 0);

  // 7) x1 = LN(y)
  layer_norm<<<dim3(MROWS), dim3(256), 0, stream>>>(ybuf, ln1_g, ln1_b, x1f, x1_bf, 4096);

  // 8) ff = relu(x1 @ W_ff1 + b_ff1) (bf16 out)
  gemm(x1_bf, Wff1_t, b_ff1, nullptr, nullptr, nullptr, ff_bf, MROWS, 2048, 4096, 1);

  // 9) y = x1 + ff @ W_ff2 + b_ff2
  gemm(ff_bf, Wff2_t, b_ff2, x1f, nullptr, ybuf, nullptr, MROWS, 4096, 2048, 0);

  // 10) union_ctx = LN(y) (bf16 only)
  layer_norm<<<dim3(MROWS), dim3(256), 0, stream>>>(ybuf, ln2_g, ln2_b, nullptr, uc_bf, 4096);

  // 11) visual = (union_ctx @ W_peu + b_peu) * ctx_gate (in place)
  gemm(uc_bf, Wpeu_t, b_peu, nullptr, ctx_gate, ctx_gate, nullptr,
       MROWS, 4096, 4096, 0);

  // 12) output head
  final_proj<<<dim3(MROWS), dim3(256), 0, stream>>>(
      ctx_gate, prod_f, W_rel, b_rel, W_ctx, b_ctx, freq, pairs, obj_preds, out);
}
